// MoDBlock_33028298506455
// MI455X (gfx1250) — compile-verified, matrix operand order checked by reading, not run on hardware
//
#include <hip/hip_runtime.h>

// ---------------------------------------------------------------------------
// MoD transformer block for gfx1250 (MI455X). bf16 WMMA for every GEMM
// (QKV / O-proj / gate / up / down + flash-attention QK^T and PV), with
// CDNA5 async global->LDS staging (ASYNCcnt) for the matmul tiles.
// ---------------------------------------------------------------------------

typedef __attribute__((ext_vector_type(16))) __bf16 v16bf;
typedef __attribute__((ext_vector_type(8)))  float  v8f;

constexpr int kB  = 2;
constexpr int kT  = 4096;
constexpr int kD  = 1024;
constexpr int kH  = 16;
constexpr int kHD = 64;
constexpr int kC  = 2048;          // selected tokens per batch (CAP=0.5)
constexpr int kFF = 4096;
constexpr int kBC = kB * kC;       // 4096 selected rows total

// ---------------------------------------------------------------------------
// WMMA helpers. CDNA5 16-bit A-operand layout (ISA 7.12.2): lanes 0-15 hold
// M=lane, K={0..7,16..23}; lanes 16-31 hold M=lane-16, K={8..15,24..31};
// two bf16 per VGPR. With row-major (M x K) data that is two contiguous
// 16-byte chunks per lane. Every B operand is kept pre-transposed (N x K
// row-major) so B fragments use the identical load pattern.
// ---------------------------------------------------------------------------
union FragU { uint4 u[2]; v16bf v; };

__device__ inline v16bf ld_frag(const __bf16* row, int lh) {
  FragU f;
  const __bf16* p = row + (lh ? 8 : 0);
  f.u[0] = *(const uint4*)p;
  f.u[1] = *(const uint4*)(p + 16);
  return f.v;
}

__device__ inline v8f wmma_bf16(v16bf a, v16bf b, v8f c) {
  // D(16x16,f32) = A(16x32,bf16) * B(32x16,bf16) + C
  return __builtin_amdgcn_wmma_f32_16x16x32_bf16(false, a, false, b,
                                                 (short)0, c, false, false);
}

// ---------------------------------------------------------------------------
// CDNA5 async DMA: global -> LDS, 16 bytes, tracked by ASYNCcnt (no VGPR
// round-trip). LDS address = low 32 bits of the generic shared pointer.
// ---------------------------------------------------------------------------
__device__ inline void async_copy_b128(const void* g, void* lds) {
  const unsigned l = (unsigned)(unsigned long long)lds;
  asm volatile("global_load_async_to_lds_b128 %0, %1, off"
               :: "v"(l), "v"(g)
               : "memory");
}
__device__ inline void async_wait0() {
  asm volatile("s_wait_asynccnt 0x0" ::: "memory");
}

// ---------------------------------------------------------------------------
// Weight convert + transpose: in (K x N, f32) -> out (N x K, bf16)
// ---------------------------------------------------------------------------
__global__ __launch_bounds__(256) void k_transpose_bf16(
    const float* __restrict__ in, __bf16* __restrict__ out, int K, int N) {
  __shared__ float tile[32][33];
  const int n0 = blockIdx.x * 32, k0 = blockIdx.y * 32;
  const int tx = threadIdx.x & 31, ty = threadIdx.x >> 5;  // 32 x 8
#pragma unroll
  for (int i = 0; i < 32; i += 8)
    tile[ty + i][tx] = in[(long)(k0 + ty + i) * N + n0 + tx];
  __syncthreads();
#pragma unroll
  for (int i = 0; i < 32; i += 8)
    out[(long)(n0 + ty + i) * K + k0 + tx] = (__bf16)tile[tx][ty + i];
}

// ---------------------------------------------------------------------------
// Router: per-token scalar logit, one wave32 per token.
// ---------------------------------------------------------------------------
__global__ __launch_bounds__(256) void k_router(
    const float* __restrict__ x, const float* __restrict__ wr,
    const float* __restrict__ br, float* __restrict__ scores) {
  const int token = blockIdx.x * 8 + (threadIdx.x >> 5);
  const int lane  = threadIdx.x & 31;
  const float* row = x + (long)token * kD;
  float s = 0.f;
#pragma unroll 8
  for (int j = lane; j < kD; j += 32) s += row[j] * wr[j];
#pragma unroll
  for (int off = 16; off > 0; off >>= 1) s += __shfl_xor(s, off, 32);
  if (lane == 0) scores[token] = s + br[0];
}

// ---------------------------------------------------------------------------
// Top-C per batch: bitonic sort of (score-desc, idx-asc) keys in LDS, take
// first C indices, bitonic sort them ascending (matches top_k + sort).
// ---------------------------------------------------------------------------
__global__ __launch_bounds__(1024) void k_topk(
    const float* __restrict__ scores, unsigned* __restrict__ idx_out) {
  __shared__ unsigned long long key[kT];
  __shared__ unsigned sidx[kC];
  const int b = blockIdx.x, tid = threadIdx.x;
  for (int i = tid; i < kT; i += 1024) {
    unsigned u   = __float_as_uint(scores[b * kT + i]);
    unsigned ord = (u & 0x80000000u) ? ~u : (u | 0x80000000u);  // asc order
    key[i] = ((unsigned long long)(~ord) << 32) | (unsigned)i;  // desc score
  }
  __syncthreads();
  for (int k = 2; k <= kT; k <<= 1)
    for (int j = k >> 1; j > 0; j >>= 1) {
      for (int i = tid; i < kT; i += 1024) {
        int p = i ^ j;
        if (p > i) {
          bool up = (i & k) == 0;
          unsigned long long a = key[i], c = key[p];
          if ((a > c) == up) { key[i] = c; key[p] = a; }
        }
      }
      __syncthreads();
    }
  for (int i = tid; i < kC; i += 1024)
    sidx[i] = (unsigned)(key[i] & 0xFFFFFFFFull);
  __syncthreads();
  for (int k = 2; k <= kC; k <<= 1)
    for (int j = k >> 1; j > 0; j >>= 1) {
      for (int i = tid; i < kC; i += 1024) {
        int p = i ^ j;
        if (p > i) {
          bool up = (i & k) == 0;
          unsigned a = sidx[i], c = sidx[p];
          if ((a > c) == up) { sidx[i] = c; sidx[p] = a; }
        }
      }
      __syncthreads();
    }
  for (int i = tid; i < kC; i += 1024) idx_out[b * kC + i] = sidx[i];
}

// ---------------------------------------------------------------------------
// Gather selected tokens + RMSNorm(g1): sel (f32) and xn (bf16). 1 block/row.
// ---------------------------------------------------------------------------
__global__ __launch_bounds__(256) void k_gather_norm(
    const float* __restrict__ x, const unsigned* __restrict__ idx,
    const float* __restrict__ g, float* __restrict__ sel,
    __bf16* __restrict__ xn) {
  const int r = blockIdx.x;
  const int b = r >> 11;  // /kC
  const unsigned tok = idx[r];
  const float* src = x + ((long)b * kT + tok) * kD;
  const int base = threadIdx.x * 4;
  float4 f = *(const float4*)(src + base);
  float ss = f.x * f.x + f.y * f.y + f.z * f.z + f.w * f.w;
#pragma unroll
  for (int off = 16; off > 0; off >>= 1) ss += __shfl_xor(ss, off, 32);
  __shared__ float red[8];
  if ((threadIdx.x & 31) == 0) red[threadIdx.x >> 5] = ss;
  __syncthreads();
  float tot = 0.f;
#pragma unroll
  for (int i = 0; i < 8; ++i) tot += red[i];
  const float inv = rsqrtf(tot * (1.f / kD) + 1e-6f);
  *(float4*)(sel + (long)r * kD + base) = f;
  __bf16* xo = xn + (long)r * kD + base;
  const float* gp = g + base;
  xo[0] = (__bf16)(f.x * inv * gp[0]);
  xo[1] = (__bf16)(f.y * inv * gp[1]);
  xo[2] = (__bf16)(f.z * inv * gp[2]);
  xo[3] = (__bf16)(f.w * inv * gp[3]);
}

// ---------------------------------------------------------------------------
// Residual + RMSNorm(g2): r1 = sel + hat (f32), z = rmsnorm(r1) (bf16).
// ---------------------------------------------------------------------------
__global__ __launch_bounds__(256) void k_resid_norm(
    const float* __restrict__ sel, const float* __restrict__ hat,
    const float* __restrict__ g, float* __restrict__ r1,
    __bf16* __restrict__ z) {
  const int r = blockIdx.x;
  const int base = threadIdx.x * 4;
  const long o = (long)r * kD + base;
  float4 a = *(const float4*)(sel + o);
  float4 hb = *(const float4*)(hat + o);
  a.x += hb.x; a.y += hb.y; a.z += hb.z; a.w += hb.w;
  float ss = a.x * a.x + a.y * a.y + a.z * a.z + a.w * a.w;
#pragma unroll
  for (int off = 16; off > 0; off >>= 1) ss += __shfl_xor(ss, off, 32);
  __shared__ float red[8];
  if ((threadIdx.x & 31) == 0) red[threadIdx.x >> 5] = ss;
  __syncthreads();
  float tot = 0.f;
#pragma unroll
  for (int i = 0; i < 8; ++i) tot += red[i];
  const float inv = rsqrtf(tot * (1.f / kD) + 1e-6f);
  *(float4*)(r1 + o) = a;
  const float* gp = g + base;
  __bf16* zo = z + o;
  zo[0] = (__bf16)(a.x * inv * gp[0]);
  zo[1] = (__bf16)(a.y * inv * gp[1]);
  zo[2] = (__bf16)(a.z * inv * gp[2]);
  zo[3] = (__bf16)(a.w * inv * gp[3]);
}

// ---------------------------------------------------------------------------
// WMMA GEMM: C(MxN) = A(MxK,bf16 row-major) * BT(NxK,bf16 row-major)^T.
// Block 256 = 8 waves in a 2x4 grid; block tile 128x256; wave tile 64x64
// (4x4 f32 accumulators -> 16 WMMA per 16 ds_read_b128 per K-step).
// Tiles staged with async global->LDS DMA. Dims are multiples of the tile.
// ---------------------------------------------------------------------------
template <typename OutT>
__global__ __launch_bounds__(256) void k_gemm_bf16(
    const __bf16* __restrict__ A, const __bf16* __restrict__ BT,
    OutT* __restrict__ Cmat, int M, int N, int K) {
  __shared__ __align__(16) __bf16 As[128][32];   //  8 KB
  __shared__ __align__(16) __bf16 Bs[256][32];   // 16 KB
  const int m0 = blockIdx.y * 128;
  const int n0 = blockIdx.x * 256;
  const int wave = threadIdx.x >> 5, lane = threadIdx.x & 31;
  const int lh = lane >> 4, ll = lane & 15;
  const int wm = wave & 1, wn = wave >> 1;  // 2 x 4 wave grid
  v8f acc[4][4] = {};
  for (int k0 = 0; k0 < K; k0 += 32) {
    __syncthreads();
    // A tile: 512 x 16B chunks; B tile: 1024 x 16B chunks; 256 threads.
#pragma unroll
    for (int i = 0; i < 2; ++i) {
      const int c = threadIdx.x + i * 256;
      const int r = c >> 2, e = (c & 3) * 8;
      async_copy_b128(A + (long)(m0 + r) * K + k0 + e, &As[r][e]);
    }
#pragma unroll
    for (int i = 0; i < 4; ++i) {
      const int c = threadIdx.x + i * 256;
      const int r = c >> 2, e = (c & 3) * 8;
      async_copy_b128(BT + (long)(n0 + r) * K + k0 + e, &Bs[r][e]);
    }
    async_wait0();
    __syncthreads();
    v16bf bf[4];
#pragma unroll
    for (int nt = 0; nt < 4; ++nt)
      bf[nt] = ld_frag(&Bs[wn * 64 + nt * 16 + ll][0], lh);
#pragma unroll
    for (int mt = 0; mt < 4; ++mt) {
      const v16bf af = ld_frag(&As[wm * 64 + mt * 16 + ll][0], lh);
#pragma unroll
      for (int nt = 0; nt < 4; ++nt)
        acc[mt][nt] = wmma_bf16(af, bf[nt], acc[mt][nt]);
    }
  }
#pragma unroll
  for (int mt = 0; mt < 4; ++mt)
#pragma unroll
    for (int nt = 0; nt < 4; ++nt)
#pragma unroll
      for (int r = 0; r < 8; ++r) {
        const int row = m0 + wm * 64 + mt * 16 + r + 8 * lh;
        const int col = n0 + wn * 64 + nt * 16 + ll;
        Cmat[(long)row * N + col] = (OutT)acc[mt][nt][r];
      }
}

// ---------------------------------------------------------------------------
// Flash attention over selected tokens. Block 128 = 4 waves; block q-tile 64
// (16 q rows per wave); 64-key tiles staged in LDS (K via async DMA,
// V transposed on store so the PV B operand is contiguous per fragment).
// ---------------------------------------------------------------------------
__global__ __launch_bounds__(128) void k_attention(
    const __bf16* __restrict__ Q, const __bf16* __restrict__ K,
    const __bf16* __restrict__ V, __bf16* __restrict__ O) {
  const int qt = blockIdx.x;   // q tile (64 rows)
  const int h  = blockIdx.y;
  const int b  = blockIdx.z;
  const int wave = threadIdx.x >> 5;
  const int lane = threadIdx.x & 31;
  const int lh = lane >> 4, ll = lane & 15;

  __shared__ __align__(16) __bf16 Ks[64][64];
  __shared__ __align__(16) __bf16 Vts[64][64];      // [hd][token]
  __shared__ __align__(16) __bf16 Ps[4][16][64];    // per-wave P tile

  // Q fragments for this wave's 16 rows, loaded once from global.
  const long qtok = (long)b * kC + qt * 64 + wave * 16 + ll;
  const __bf16* qp = Q + qtok * kD + h * kHD;
  const v16bf qfrag0 = ld_frag(qp, lh);
  const v16bf qfrag1 = ld_frag(qp + 32, lh);

  v8f acc_o[4] = {};
  float row_m[8], row_l[8];
#pragma unroll
  for (int r = 0; r < 8; ++r) { row_m[r] = -3.0e38f; row_l[r] = 0.f; }
  const float scale = 0.125f;  // 1/sqrt(64)

  for (int kt = 0; kt <= qt; ++kt) {
    __syncthreads();  // previous tile fully consumed
#pragma unroll
    for (int i = 0; i < 4; ++i) {
      const int c = threadIdx.x + i * 128;
      const int r = c >> 3;
      const int e = (c & 7) * 8;
      const long tok = (long)b * kC + kt * 64 + r;
      async_copy_b128(K + tok * kD + h * kHD + e, &Ks[r][e]);
      union { uint4 u; __bf16 el[8]; } tv;
      tv.u = *(const uint4*)(V + tok * kD + h * kHD + e);
#pragma unroll
      for (int j = 0; j < 8; ++j) Vts[e + j][r] = tv.el[j];
    }
    async_wait0();
    __syncthreads();

    // S = Q K^T (16 q x 64 k), f32 accum
    v8f acc_s[4] = {};
#pragma unroll
    for (int nt = 0; nt < 4; ++nt) {
      const v16bf kf0 = ld_frag(&Ks[nt * 16 + ll][0], lh);
      const v16bf kf1 = ld_frag(&Ks[nt * 16 + ll][32], lh);
      acc_s[nt] = wmma_bf16(qfrag0, kf0, acc_s[nt]);
      acc_s[nt] = wmma_bf16(qfrag1, kf1, acc_s[nt]);
    }

    // scale + causal mask (select, no divergence)
    const int qg = qt * 64 + wave * 16 + 8 * lh;
#pragma unroll
    for (int nt = 0; nt < 4; ++nt) {
      const int kg = kt * 64 + nt * 16 + ll;
#pragma unroll
      for (int r = 0; r < 8; ++r) {
        const float s = acc_s[nt][r] * scale;
        acc_s[nt][r] = (kg <= qg + r) ? s : -1.0e30f;
      }
    }

    // streaming softmax (row reductions across the 16-lane half-groups)
#pragma unroll
    for (int r = 0; r < 8; ++r) {
      float m = fmaxf(fmaxf(acc_s[0][r], acc_s[1][r]),
                      fmaxf(acc_s[2][r], acc_s[3][r]));
#pragma unroll
      for (int off = 1; off < 16; off <<= 1) m = fmaxf(m, __shfl_xor(m, off, 32));
      const float mo = row_m[r];
      const float mn = fmaxf(mo, m);
      row_m[r] = mn;
      const float sf = __expf(mo - mn);
      row_l[r] *= sf;
#pragma unroll
      for (int nt = 0; nt < 4; ++nt) acc_o[nt][r] *= sf;
      float rs = 0.f;
#pragma unroll
      for (int nt = 0; nt < 4; ++nt) {
        const float pv = __expf(acc_s[nt][r] - mn);
        acc_s[nt][r] = pv;
        rs += pv;
      }
#pragma unroll
      for (int off = 1; off < 16; off <<= 1) rs += __shfl_xor(rs, off, 32);
      row_l[r] += rs;
    }

    // P: C/D layout -> bf16 in per-wave LDS -> reload as A fragments
#pragma unroll
    for (int nt = 0; nt < 4; ++nt)
#pragma unroll
      for (int r = 0; r < 8; ++r)
        Ps[wave][r + 8 * lh][nt * 16 + ll] = (__bf16)acc_s[nt][r];

    const v16bf pf0 = ld_frag(&Ps[wave][ll][0], lh);
    const v16bf pf1 = ld_frag(&Ps[wave][ll][32], lh);
#pragma unroll
    for (int nt = 0; nt < 4; ++nt) {
      const v16bf vf0 = ld_frag(&Vts[nt * 16 + ll][0], lh);
      const v16bf vf1 = ld_frag(&Vts[nt * 16 + ll][32], lh);
      acc_o[nt] = wmma_bf16(pf0, vf0, acc_o[nt]);
      acc_o[nt] = wmma_bf16(pf1, vf1, acc_o[nt]);
    }
  }

  const long otok = (long)b * kC + qt * 64 + wave * 16;
#pragma unroll
  for (int r = 0; r < 8; ++r) {
    const float inv = 1.f / row_l[r];
    const long row = otok + r + 8 * lh;
#pragma unroll
    for (int nt = 0; nt < 4; ++nt)
      O[row * kD + h * kHD + nt * 16 + ll] = (__bf16)(acc_o[nt][r] * inv);
  }
}

// ---------------------------------------------------------------------------
// act = silu(gate) * up, in place in gate buffer; 8 bf16 per thread.
// ---------------------------------------------------------------------------
__global__ __launch_bounds__(256) void k_swiglu(
    __bf16* __restrict__ gate, const __bf16* __restrict__ up) {
  const long i = ((long)blockIdx.x * 256 + threadIdx.x) * 8;
  union U8 { uint4 u; __bf16 e[8]; } gg, uu;
  gg.u = *(const uint4*)(gate + i);
  uu.u = *(const uint4*)(up + i);
#pragma unroll
  for (int j = 0; j < 8; ++j) {
    const float g = (float)gg.e[j];
    const float u = (float)uu.e[j];
    gg.e[j] = (__bf16)((g / (1.f + __expf(-g))) * u);
  }
  *(uint4*)(gate + i) = gg.u;
}

// ---------------------------------------------------------------------------
// Scatter back: out[b, idx[c], :] = r1 + mlp_out (out already holds x).
// ---------------------------------------------------------------------------
__global__ __launch_bounds__(256) void k_scatter(
    const unsigned* __restrict__ idx, const float* __restrict__ r1,
    const float* __restrict__ mlp, float* __restrict__ out) {
  const int r = blockIdx.x;
  const int b = r >> 11;  // /kC
  const unsigned tok = idx[r];
  const int base = threadIdx.x * 4;
  const long o = (long)r * kD + base;
  float4 a = *(const float4*)(r1 + o);
  const float4 c = *(const float4*)(mlp + o);
  a.x += c.x; a.y += c.y; a.z += c.z; a.w += c.w;
  *(float4*)(out + ((long)b * kT + tok) * kD + base) = a;
}

// ---------------------------------------------------------------------------
extern "C" void kernel_launch(void* const* d_in, const int* in_sizes, int n_in,
                              void* d_out, int out_size, void* d_ws,
                              size_t ws_size, hipStream_t stream) {
  const float* x  = (const float*)d_in[0];
  const float* wr = (const float*)d_in[1];
  const float* br = (const float*)d_in[2];
  const float* g1 = (const float*)d_in[3];
  const float* g2 = (const float*)d_in[4];
  const float* wq = (const float*)d_in[5];
  const float* wk = (const float*)d_in[6];
  const float* wv = (const float*)d_in[7];
  const float* wo = (const float*)d_in[8];
  const float* wg = (const float*)d_in[9];
  const float* wu = (const float*)d_in[10];
  const float* wd = (const float*)d_in[11];
  float* out = (float*)d_out;

  char* p = (char*)d_ws;
  auto carve = [&](size_t bytes) -> char* {
    char* r = p;
    p += (bytes + 255) & ~(size_t)255;
    return r;
  };
  __bf16* wqT  = (__bf16*)carve((size_t)kD * kD * 2);
  __bf16* wkT  = (__bf16*)carve((size_t)kD * kD * 2);
  __bf16* wvT  = (__bf16*)carve((size_t)kD * kD * 2);
  __bf16* woT  = (__bf16*)carve((size_t)kD * kD * 2);
  __bf16* wgT  = (__bf16*)carve((size_t)kFF * kD * 2);
  __bf16* wuT  = (__bf16*)carve((size_t)kFF * kD * 2);
  __bf16* wdT  = (__bf16*)carve((size_t)kD * kFF * 2);
  float*  scores = (float*)carve((size_t)kB * kT * 4);
  unsigned* idx  = (unsigned*)carve((size_t)kB * kC * 4);
  float*  sel  = (float*)carve((size_t)kBC * kD * 4);
  __bf16* xn   = (__bf16*)carve((size_t)kBC * kD * 2);
  __bf16* Qb   = (__bf16*)carve((size_t)kBC * kD * 2);
  __bf16* Kb   = (__bf16*)carve((size_t)kBC * kD * 2);
  __bf16* Vb   = (__bf16*)carve((size_t)kBC * kD * 2);
  __bf16* attn = (__bf16*)carve((size_t)kBC * kD * 2);
  float*  hat  = (float*)carve((size_t)kBC * kD * 4);
  float*  r1   = (float*)carve((size_t)kBC * kD * 4);
  __bf16* z    = (__bf16*)carve((size_t)kBC * kD * 2);
  __bf16* gate = (__bf16*)carve((size_t)kBC * kFF * 2);
  __bf16* up   = (__bf16*)carve((size_t)kBC * kFF * 2);
  float*  mlp  = (float*)carve((size_t)kBC * kD * 4);

  // 1. weight convert + transpose to bf16 (N x K)
  k_transpose_bf16<<<dim3(kD / 32, kD / 32), 256, 0, stream>>>(wq, wqT, kD, kD);
  k_transpose_bf16<<<dim3(kD / 32, kD / 32), 256, 0, stream>>>(wk, wkT, kD, kD);
  k_transpose_bf16<<<dim3(kD / 32, kD / 32), 256, 0, stream>>>(wv, wvT, kD, kD);
  k_transpose_bf16<<<dim3(kD / 32, kD / 32), 256, 0, stream>>>(wo, woT, kD, kD);
  k_transpose_bf16<<<dim3(kFF / 32, kD / 32), 256, 0, stream>>>(wg, wgT, kD, kFF);
  k_transpose_bf16<<<dim3(kFF / 32, kD / 32), 256, 0, stream>>>(wu, wuT, kD, kFF);
  k_transpose_bf16<<<dim3(kD / 32, kFF / 32), 256, 0, stream>>>(wd, wdT, kFF, kD);

  // 2. router + top-C selection
  k_router<<<kB * kT / 8, 256, 0, stream>>>(x, wr, br, scores);
  k_topk<<<kB, 1024, 0, stream>>>(scores, idx);

  // 3. out = x (pass-through for unselected tokens)
  hipMemcpyAsync(out, x, (size_t)kB * kT * kD * sizeof(float),
                 hipMemcpyDeviceToDevice, stream);

  // 4. gather + rmsnorm(g1)
  k_gather_norm<<<kBC, 256, 0, stream>>>(x, idx, g1, sel, xn);

  // 5. Q,K,V projections (WMMA)
  k_gemm_bf16<__bf16><<<dim3(kD / 256, kBC / 128), 256, 0, stream>>>(
      xn, wqT, Qb, kBC, kD, kD);
  k_gemm_bf16<__bf16><<<dim3(kD / 256, kBC / 128), 256, 0, stream>>>(
      xn, wkT, Kb, kBC, kD, kD);
  k_gemm_bf16<__bf16><<<dim3(kD / 256, kBC / 128), 256, 0, stream>>>(
      xn, wvT, Vb, kBC, kD, kD);

  // 6. causal flash attention (WMMA)
  k_attention<<<dim3(kC / 64, kH, kB), 128, 0, stream>>>(Qb, Kb, Vb, attn);

  // 7. output projection (WMMA, f32 out)
  k_gemm_bf16<float><<<dim3(kD / 256, kBC / 128), 256, 0, stream>>>(
      attn, woT, hat, kBC, kD, kD);

  // 8. residual + rmsnorm(g2)
  k_resid_norm<<<kBC, 256, 0, stream>>>(sel, hat, g2, r1, z);

  // 9. SwiGLU MLP (WMMA)
  k_gemm_bf16<__bf16><<<dim3(kFF / 256, kBC / 128), 256, 0, stream>>>(
      z, wgT, gate, kBC, kFF, kD);
  k_gemm_bf16<__bf16><<<dim3(kFF / 256, kBC / 128), 256, 0, stream>>>(
      z, wuT, up, kBC, kFF, kD);
  k_swiglu<<<(int)((long)kBC * kFF / (256 * 8)), 256, 0, stream>>>(gate, up);
  k_gemm_bf16<float><<<dim3(kD / 256, kBC / 128), 256, 0, stream>>>(
      gate, wdT, mlp, kBC, kD, kFF);

  // 10. scatter selected rows: out[b, idx] = r1 + mlp
  k_scatter<<<kBC, 256, 0, stream>>>(idx, r1, mlp, out);
}
